// AttentionQKNorm_63410897158586
// MI455X (gfx1250) — matrix operand order repaired and checked by reading, not compiled, not run on hardware
//
#include <hip/hip_runtime.h>
#include <hip/hip_bf16.h>
#include <math.h>
#include <stdint.h>

// ---------------------------------------------------------------------------
// Types for CDNA5 WMMA (wave32): v_wmma_f32_16x16x32_bf16
// ---------------------------------------------------------------------------
typedef __attribute__((ext_vector_type(16))) __bf16 bf16x16;
typedef __attribute__((ext_vector_type(8)))  __bf16 bf16x8;
typedef __attribute__((ext_vector_type(4)))  __bf16 bf16x4;
typedef __attribute__((ext_vector_type(8)))  float  f32x8;
typedef __attribute__((ext_vector_type(4)))  float  f32x4;
typedef __attribute__((ext_vector_type(4)))  int    v4i;

union ABFrag { bf16x16 v; bf16x8 h[2]; };

#define WMMA_BF16(A, B, C) \
  __builtin_amdgcn_wmma_f32_16x16x32_bf16(false, (A), false, (B), (short)0, (C), false, false)

static constexpr int Bb   = 4;
static constexpr int Nseq = 2048;
static constexpr int Cdim = 1024;
static constexpr int Hh   = 16;
static constexpr int Dd   = 64;

// ---------------------------------------------------------------------------
// CDNA5 async global->LDS copies (ASYNCcnt path), guarded so a toolchain
// without the builtins falls back to load+ds_store.
// Builtin signature (from this toolchain's diagnostic):
//   (v4i addrspace(1)* gsrc, v4i addrspace(3)* ldst, imm offset, imm cpol)
// ---------------------------------------------------------------------------
#if defined(__has_builtin)
#if __has_builtin(__builtin_amdgcn_global_load_async_to_lds_b128) && \
    __has_builtin(__builtin_amdgcn_s_wait_asynccnt)
#define HAVE_ASYNC_LDS 1
#endif
#endif

typedef __attribute__((address_space(1))) v4i* gv4i_p;
typedef __attribute__((address_space(3))) v4i* lv4i_p;

__device__ __forceinline__ void async_copy_b128(const void* g, void* l) {
#if defined(HAVE_ASYNC_LDS)
  __builtin_amdgcn_global_load_async_to_lds_b128(
      (gv4i_p)(uintptr_t)g, (lv4i_p)(uintptr_t)l, 0, 0);
#else
  *(bf16x8*)l = *(const bf16x8*)g;
#endif
}

__device__ __forceinline__ void async_fence() {
#if defined(HAVE_ASYNC_LDS)
  __builtin_amdgcn_s_wait_asynccnt(0);
#endif
}

// ---------------------------------------------------------------------------
// Kernel 1: fp32 -> bf16 conversion (read-once data: non-temporal loads)
// ---------------------------------------------------------------------------
__global__ __launch_bounds__(256) void cvt_f32_bf16(const float* __restrict__ in,
                                                    __bf16* __restrict__ out) {
  const int i = (blockIdx.x * 256 + threadIdx.x) * 4;
  f32x4 f = __builtin_nontemporal_load((const f32x4*)(in + i));
  bf16x4 o = { (__bf16)f.x, (__bf16)f.y, (__bf16)f.z, (__bf16)f.w };
  *(bf16x4*)(out + i) = o;
}

// ---------------------------------------------------------------------------
// Shared GEMM main loop: 128x128 block tile, K = 1024 in steps of 32.
// 8 waves arranged 4(M) x 2(N); each wave computes a 32x64 sub-tile
// (2x4 fragments of 16x16) -> 8 WMMA per K-step, 256 WMMA per wave total,
// fed by 12 ds_load_b128 per K-step.
// A tile staged row-major (async copy when available), B tile staged
// TRANSPOSED (pitch 40 halfs) so fragment loads are contiguous 16B.
// ---------------------------------------------------------------------------
__device__ __forceinline__ void gemm_tile_128x128(
    const __bf16* __restrict__ X,          // [M, 1024] bf16, row-major
    const __bf16* __restrict__ W, int ldw, // [1024, ldw] bf16, row-major
    __bf16* As, __bf16* Bs,                // LDS: As[128*40], Bs[128*40]
    int m0, int c0, f32x8 (&acc)[2][4]) {
  const int tid  = threadIdx.x;
  const int lane = tid & 31;
  const int wid  = tid >> 5;
  const int wm   = wid & 3;    // wave M index (0..3) -> 32 rows each
  const int wn   = wid >> 2;   // wave N index (0..1) -> 64 cols each
  const int lr   = lane & 15;
  const int lh   = lane >> 4;

  for (int k0 = 0; k0 < 1024; k0 += 32) {
    // --- stage A tile: 128 rows x 32 halfs, 2 x b128 per thread (async) ---
    {
      const int row = tid >> 2;
      const int off = (tid & 3) * 8;
      async_copy_b128(&X[(size_t)(m0 + row) * 1024 + k0 + off],
                      &As[row * 40 + off]);
      async_copy_b128(&X[(size_t)(m0 + row + 64) * 1024 + k0 + off],
                      &As[(row + 64) * 40 + off]);
    }
    // --- stage B tile transposed: 32(K) x 128(N) -> Bs[n][k] ---
    {
      const int kr   = tid >> 3;          // 0..31
      const int coff = (tid & 7) * 16;    // 0..112
      bf16x8 w0 = *(const bf16x8*)&W[(size_t)(k0 + kr) * ldw + c0 + coff];
      bf16x8 w1 = *(const bf16x8*)&W[(size_t)(k0 + kr) * ldw + c0 + coff + 8];
#pragma unroll
      for (int j = 0; j < 8; ++j) {
        Bs[(coff + j) * 40 + kr]     = w0[j];
        Bs[(coff + 8 + j) * 40 + kr] = w1[j];
      }
      if (k0 + 32 < 1024) {  // emits global_prefetch_b8 for next K tile
        __builtin_prefetch(&X[(size_t)(m0 + (tid >> 2)) * 1024 + k0 + 32], 0, 1);
        __builtin_prefetch(&W[(size_t)(k0 + 32 + kr) * ldw + c0 + coff], 0, 1);
      }
    }
    async_fence();
    __syncthreads();

    // --- fragment loads (contiguous k = 16*lh + s, same order for A and B) + 2x4 WMMA ---
    ABFrag af[2], bf[4];
#pragma unroll
    for (int i = 0; i < 2; ++i) {
      const int row = wm * 32 + i * 16 + lr;
      const int kk  = lh * 16;  // lanes 0-15: K 0..15; lanes 16-31: K 16..31
      af[i].h[0] = *(const bf16x8*)&As[row * 40 + kk];
      af[i].h[1] = *(const bf16x8*)&As[row * 40 + kk + 8];
    }
#pragma unroll
    for (int j = 0; j < 4; ++j) {
      const int col  = wn * 64 + j * 16 + lr;
      const int koff = lh * 16;  // lanes 0-15: K 0..15; lanes 16-31: K 16..31
      bf[j].h[0] = *(const bf16x8*)&Bs[col * 40 + koff];
      bf[j].h[1] = *(const bf16x8*)&Bs[col * 40 + koff + 8];
    }
#pragma unroll
    for (int i = 0; i < 2; ++i)
#pragma unroll
      for (int j = 0; j < 4; ++j)
        acc[i][j] = WMMA_BF16(af[i].v, bf[j].v, acc[i][j]);
    __syncthreads();
  }
}

// ---------------------------------------------------------------------------
// Kernel 2: QKV GEMM.  qkv = x @ w_qkv + b_qkv, scattered to q/k/v [B,H,N,D].
// A 128-wide output block stays inside one s (q/k/v) slice; h varies within.
// ---------------------------------------------------------------------------
__global__ __launch_bounds__(256) void gemm_qkv_kernel(
    const __bf16* __restrict__ X, const __bf16* __restrict__ W,
    const float* __restrict__ bias, __bf16* __restrict__ qb,
    __bf16* __restrict__ kb, __bf16* __restrict__ vb) {
  __shared__ __align__(16) __bf16 As[128 * 40];
  __shared__ __align__(16) __bf16 Bs[128 * 40];
  const int tid = threadIdx.x, lane = tid & 31, wid = tid >> 5;
  const int wm = wid & 3, wn = wid >> 2;
  const int lr = lane & 15, lh = lane >> 4;
  const int m0 = blockIdx.y * 128;
  const int c0 = blockIdx.x * 128;

  f32x8 acc[2][4] = {};
  gemm_tile_128x128(X, W, 3 * Cdim, As, Bs, m0, c0, acc);

  const int s = c0 >> 10;  // uniform: 1024 % 128 == 0
  __bf16* dst = (s == 0) ? qb : ((s == 1) ? kb : vb);
#pragma unroll
  for (int i = 0; i < 2; ++i)
#pragma unroll
    for (int j = 0; j < 4; ++j)
#pragma unroll
      for (int g = 0; g < 8; ++g) {
        const int row  = m0 + wm * 32 + i * 16 + g + lh * 8;
        const int c    = c0 + wn * 64 + j * 16 + lr;
        const int h    = (c >> 6) & 15;
        const int d    = c & 63;
        const int bidx = row >> 11;       // row / Nseq
        const int n    = row & 2047;      // row % Nseq
        const float v  = acc[i][j][g] + bias[c];
        dst[(((size_t)bidx * Hh + h) * Nseq + n) * Dd + d] = (__bf16)v;
      }
}

// ---------------------------------------------------------------------------
// Kernel 3: per-head LayerNorm over D=64, in place on bf16 [B,H,N,D].
// One wave per row; 2 elements / lane; f32 shuffle reductions.
// "extra" folds softmax scale D^-0.5 * log2(e) into Q.
// ---------------------------------------------------------------------------
__global__ __launch_bounds__(256) void headln_kernel(__bf16* __restrict__ t,
                                                     const float* __restrict__ gamma,
                                                     float extra) {
  const int row  = blockIdx.x * 8 + (threadIdx.x >> 5);
  const int lane = threadIdx.x & 31;
  __bf16* p = t + (size_t)row * Dd;
  float x0 = (float)p[lane * 2];
  float x1 = (float)p[lane * 2 + 1];
  float s = x0 + x1;
#pragma unroll
  for (int m = 16; m >= 1; m >>= 1) s += __shfl_xor(s, m, 32);
  const float mu = s * (1.0f / 64.0f);
  const float d0 = x0 - mu, d1 = x1 - mu;
  float v = d0 * d0 + d1 * d1;
#pragma unroll
  for (int m = 16; m >= 1; m >>= 1) v += __shfl_xor(v, m, 32);
  const float w = rsqrtf(v * (1.0f / 64.0f) + 1e-6f);
  p[lane * 2]     = (__bf16)(d0 * w * gamma[lane * 2] * extra);
  p[lane * 2 + 1] = (__bf16)(d1 * w * gamma[lane * 2 + 1] * extra);
}

// ---------------------------------------------------------------------------
// Kernel 4: flash attention.  One workgroup per (b, h, 128-query tile);
// 8 waves, each owns 16 query rows.  K/V streamed through LDS in 64-key
// tiles (K via async copy); online softmax in base-2 (scale folded into Q).
// P is routed C-layout -> A-layout through per-wave LDS staging.
// ---------------------------------------------------------------------------
__global__ __launch_bounds__(256) void attn_kernel(
    const __bf16* __restrict__ qb, const __bf16* __restrict__ kb,
    const __bf16* __restrict__ vb, __bf16* __restrict__ ob) {
  __shared__ __align__(16) __bf16 Ks[64 * 72];       // K tile, row = key
  __shared__ __align__(16) __bf16 Vt[64 * 72];       // V tile transposed, row = d
  __shared__ __align__(16) __bf16 Ps[8 * 16 * 72];   // per-wave P staging

  const int tid = threadIdx.x, lane = tid & 31, wid = tid >> 5;
  const int lr = lane & 15, lh = lane >> 4;
  const int b = blockIdx.z, h = blockIdx.y;
  const int q0 = blockIdx.x * 128;
  const size_t bh = ((size_t)b * Hh + h) * Nseq;

  // ---- load Q fragments once (contiguous in [B,H,N,D]) ----
  const __bf16* Qp = qb + (bh + q0 + wid * 16) * Dd;
  ABFrag qf[2];
#pragma unroll
  for (int c = 0; c < 2; ++c) {
    const __bf16* r = Qp + (size_t)lr * Dd + c * 32 + lh * 16;
    qf[c].h[0] = *(const bf16x8*)r;
    qf[c].h[1] = *(const bf16x8*)(r + 8);
  }

  float mrow[8], lrow[8];
#pragma unroll
  for (int g = 0; g < 8; ++g) { mrow[g] = -1e30f; lrow[g] = 0.0f; }
  f32x8 accO[4] = {};
  __bf16* Pw = &Ps[wid * 16 * 72];

  for (int kt = 0; kt < Nseq; kt += 64) {
    // ---- stage K tile (async direct copy) and V tile (transposed) ----
    {
      const int key  = tid >> 2;
      const int doff = (tid & 3) * 16;
      const __bf16* kp = kb + (bh + kt + key) * Dd + doff;
      async_copy_b128(kp,     &Ks[key * 72 + doff]);
      async_copy_b128(kp + 8, &Ks[key * 72 + doff + 8]);
      const __bf16* vp = vb + (bh + kt + key) * Dd + doff;
      bf16x8 v0 = *(const bf16x8*)vp;
      bf16x8 v1 = *(const bf16x8*)(vp + 8);
#pragma unroll
      for (int j = 0; j < 8; ++j) {
        Vt[(doff + j) * 72 + key]     = v0[j];
        Vt[(doff + 8 + j) * 72 + key] = v1[j];
      }
    }
    async_fence();
    __syncthreads();

    // ---- S = Q @ K^T : 4 key sub-tiles x 2 K-chunks = 8 WMMA ----
    f32x8 accS[4] = {};
#pragma unroll
    for (int nt = 0; nt < 4; ++nt) {
#pragma unroll
      for (int c = 0; c < 2; ++c) {
        ABFrag kf;
        const int key = nt * 16 + lr;
        const int off = c * 32 + lh * 16;
        kf.h[0] = *(const bf16x8*)&Ks[key * 72 + off];
        kf.h[1] = *(const bf16x8*)&Ks[key * 72 + off + 8];
        accS[nt] = WMMA_BF16(qf[c].v, kf.v, accS[nt]);
      }
    }

    // ---- online softmax (base-2) with 16-lane row reductions ----
    float alpha[8];
#pragma unroll
    for (int g = 0; g < 8; ++g) {
      float mx = fmaxf(fmaxf(accS[0][g], accS[1][g]),
                       fmaxf(accS[2][g], accS[3][g]));
#pragma unroll
      for (int m = 8; m >= 1; m >>= 1) mx = fmaxf(mx, __shfl_xor(mx, m, 32));
      const float mnew = fmaxf(mrow[g], mx);
      alpha[g] = exp2f(mrow[g] - mnew);
      float rs = 0.0f;
#pragma unroll
      for (int nt = 0; nt < 4; ++nt) {
        const float p = exp2f(accS[nt][g] - mnew);
        accS[nt][g] = p;
        rs += p;
      }
#pragma unroll
      for (int m = 8; m >= 1; m >>= 1) rs += __shfl_xor(rs, m, 32);
      lrow[g] = lrow[g] * alpha[g] + rs;
      mrow[g] = mnew;
#pragma unroll
      for (int nt = 0; nt < 4; ++nt) accO[nt][g] *= alpha[g];
    }

    // ---- stage P: C-layout -> LDS (row-major 16 x 64) ----
#pragma unroll
    for (int nt = 0; nt < 4; ++nt)
#pragma unroll
      for (int g = 0; g < 8; ++g)
        Pw[(g + lh * 8) * 72 + nt * 16 + lr] = (__bf16)accS[nt][g];
    __syncthreads();

    // ---- O += P @ V : 2 key-chunks x 4 d sub-tiles = 8 WMMA ----
#pragma unroll
    for (int c = 0; c < 2; ++c) {
      ABFrag pf;
      const int kk = lh * 16;
      pf.h[0] = *(const bf16x8*)&Pw[lr * 72 + c * 32 + kk];
      pf.h[1] = *(const bf16x8*)&Pw[lr * 72 + c * 32 + kk + 8];
#pragma unroll
      for (int nt = 0; nt < 4; ++nt) {
        ABFrag vf;
        const int drow = nt * 16 + lr;
        const int off  = c * 32 + lh * 16;
        vf.h[0] = *(const bf16x8*)&Vt[drow * 72 + off];
        vf.h[1] = *(const bf16x8*)&Vt[drow * 72 + off + 8];
        accO[nt] = WMMA_BF16(pf.v, vf.v, accO[nt]);
      }
    }
    __syncthreads();
  }

  // ---- epilogue: O /= l ; write bf16 into [B,N,C] at column h*64 ----
#pragma unroll
  for (int g = 0; g < 8; ++g) {
    const float inv = __builtin_amdgcn_rcpf(lrow[g]);
    const int row = q0 + wid * 16 + g + lh * 8;
    __bf16* op = ob + ((size_t)b * Nseq + row) * Cdim + h * Dd;
#pragma unroll
    for (int nt = 0; nt < 4; ++nt)
      op[nt * 16 + lr] = (__bf16)(accO[nt][g] * inv);
  }
}

// ---------------------------------------------------------------------------
// Kernel 5: output projection.  out = o @ w_proj + b_proj (fp32 out).
// ---------------------------------------------------------------------------
__global__ __launch_bounds__(256) void gemm_proj_kernel(
    const __bf16* __restrict__ A, const __bf16* __restrict__ W,
    const float* __restrict__ bias, float* __restrict__ out) {
  __shared__ __align__(16) __bf16 As[128 * 40];
  __shared__ __align__(16) __bf16 Bs[128 * 40];
  const int tid = threadIdx.x, lane = tid & 31, wid = tid >> 5;
  const int wm = wid & 3, wn = wid >> 2;
  const int lr = lane & 15, lh = lane >> 4;
  const int m0 = blockIdx.y * 128;
  const int c0 = blockIdx.x * 128;

  f32x8 acc[2][4] = {};
  gemm_tile_128x128(A, W, Cdim, As, Bs, m0, c0, acc);

#pragma unroll
  for (int i = 0; i < 2; ++i)
#pragma unroll
    for (int j = 0; j < 4; ++j)
#pragma unroll
      for (int g = 0; g < 8; ++g) {
        const int row = m0 + wm * 32 + i * 16 + g + lh * 8;
        const int cc  = c0 + wn * 64 + j * 16 + lr;
        out[(size_t)row * Cdim + cc] = acc[i][j][g] + bias[cc];
      }
}

// ---------------------------------------------------------------------------
// Host launcher
// ---------------------------------------------------------------------------
extern "C" void kernel_launch(void* const* d_in, const int* in_sizes, int n_in,
                              void* d_out, int out_size, void* d_ws, size_t ws_size,
                              hipStream_t stream) {
  (void)in_sizes; (void)n_in; (void)out_size; (void)ws_size;
  const float* x      = (const float*)d_in[0];
  const float* w_qkv  = (const float*)d_in[1];
  const float* b_qkv  = (const float*)d_in[2];
  const float* g_q    = (const float*)d_in[3];
  const float* g_k    = (const float*)d_in[4];
  const float* w_proj = (const float*)d_in[5];
  const float* b_proj = (const float*)d_in[6];
  float* out = (float*)d_out;

  const size_t n_x     = (size_t)Bb * Nseq * Cdim;      // 8,388,608
  const size_t n_wqkv  = (size_t)Cdim * 3 * Cdim;       // 3,145,728
  const size_t n_wproj = (size_t)Cdim * Cdim;           // 1,048,576
  const size_t n_hd    = (size_t)Bb * Hh * Nseq * Dd;   // 8,388,608

  char* ws = (char*)d_ws;
  __bf16* xb     = (__bf16*)ws;                 ws += n_x * 2;
  __bf16* wqkvb  = (__bf16*)ws;                 ws += n_wqkv * 2;
  __bf16* wprojb = (__bf16*)ws;                 ws += n_wproj * 2;
  __bf16* qb     = (__bf16*)ws;                 ws += n_hd * 2;
  __bf16* kb     = (__bf16*)ws;                 ws += n_hd * 2;
  __bf16* vb     = (__bf16*)ws;                 ws += n_hd * 2;
  __bf16* ob     = (__bf16*)ws;                 ws += n_x * 2;   // [B,N,C]

  // 1) fp32 -> bf16
  cvt_f32_bf16<<<(unsigned)(n_x / 1024),     256, 0, stream>>>(x,      xb);
  cvt_f32_bf16<<<(unsigned)(n_wqkv / 1024),  256, 0, stream>>>(w_qkv,  wqkvb);
  cvt_f32_bf16<<<(unsigned)(n_wproj / 1024), 256, 0, stream>>>(w_proj, wprojb);

  // 2) QKV GEMM (8192 x 3072 x 1024) -> q/k/v [B,H,N,D]
  gemm_qkv_kernel<<<dim3(3 * Cdim / 128, Bb * Nseq / 128), 256, 0, stream>>>(
      xb, wqkvb, b_qkv, qb, kb, vb);

  // 3) per-head LayerNorm; Q gets D^-0.5 * log2(e) folded in for exp2 softmax
  const unsigned ln_blocks = (unsigned)(Bb * Hh * Nseq / 8);
  headln_kernel<<<ln_blocks, 256, 0, stream>>>(qb, g_q, 0.125f * 1.44269504088896f);
  headln_kernel<<<ln_blocks, 256, 0, stream>>>(kb, g_k, 1.0f);

  // 4) flash attention -> ob [B,N,C] bf16
  attn_kernel<<<dim3(Nseq / 128, Hh, Bb), 256, 0, stream>>>(qb, kb, vb, ob);

  // 5) output projection -> d_out fp32
  gemm_proj_kernel<<<dim3(Cdim / 128, Bb * Nseq / 128), 256, 0, stream>>>(
      ob, wprojb, b_proj, out);
}
